// GraphTransformerModel_26190710571649
// MI455X (gfx1250) — matrix operand order repaired and checked by reading, not compiled, not run on hardware
//
#include <hip/hip_runtime.h>
#include <hip/hip_bf16.h>
#include <hip/hip_fp16.h>

// Problem constants (match reference)
#define NN   100000      // nodes
#define EE   1600000     // edges
#define GG   128         // graphs
#define H    16          // hidden
#define NHD  8           // heads
#define HDIM 2           // head dim
#define LL   9           // layers
#define PED  8           // pos enc dim
#define EPSV 1e-5f

typedef __attribute__((ext_vector_type(16))) _Float16 v16h;
typedef __attribute__((ext_vector_type(8)))  _Float16 v8h;
typedef __attribute__((ext_vector_type(8)))  float    v8f;

// ---------------------------------------------------------------- helpers ---

__device__ __forceinline__ v8f wmma16(v16h a, v16h b, v8f c) {
    // D = A(16x32 f16) * B(32x16 f16) + C(16x16 f32)
    return __builtin_amdgcn_wmma_f32_16x16x32_f16(
        /*neg_a=*/false, a, /*neg_b=*/false, b,
        /*c_mod=*/(short)0, c, /*reuse_a=*/false, /*reuse_b=*/false);
}

__device__ __forceinline__ v8f bcastC(float v) {
    v8f c;
#pragma unroll
    for (int i = 0; i < 8; ++i) c[i] = v;
    return c;
}

// A fragment for K=16 real (zero-padded to 32). Lane r/half layout per ISA:
// half=0 lanes hold K={0..7,16..23}; half=1 lanes hold K={8..15,24..31}.
// Only K<16 is nonzero -> each lane loads 8 contiguous floats of its row.
__device__ __forceinline__ v16h makeA_k16(const float* row_half8, bool valid) {
    v16h a;
#pragma unroll
    for (int j = 0; j < 16; ++j) a[j] = (_Float16)0.0f;
    if (valid) {
        float4 p0 = ((const float4*)row_half8)[0];
        float4 p1 = ((const float4*)row_half8)[1];
        a[0] = (_Float16)p0.x; a[1] = (_Float16)p0.y;
        a[2] = (_Float16)p0.z; a[3] = (_Float16)p0.w;
        a[4] = (_Float16)p1.x; a[5] = (_Float16)p1.y;
        a[6] = (_Float16)p1.z; a[7] = (_Float16)p1.w;
    }
    return a;
}

// B fragment: LDS holds pre-swizzled f16 weights sB[col*32 + K] (K padded to 32).
// Lane wants B[K = half*8 + j][col] (j<8) and B[K = 16 + half*8 + (j-8)][col] (j>=8) -> two 16B reads.
__device__ __forceinline__ v16h loadB(const _Float16* sB, int col, int half) {
    v8h lo = *(const v8h*)(sB + col * 32 + half * 8);
    v8h hi = *(const v8h*)(sB + col * 32 + 16 + half * 8);
    return __builtin_shufflevector(lo, hi, 0, 1, 2, 3, 4, 5, 6, 7, 8, 9, 10, 11, 12, 13, 14, 15);
}

// Monotonic float <-> uint encoding so segment-max is ONE branchless
// global_atomic_max_u32.  encode(-inf) = 0x007FFFFF is the init value.
__device__ __forceinline__ unsigned encodeM(float f) {
    unsigned fi = __float_as_uint(f);
    return fi ^ (0x80000000u | (unsigned)((int)fi >> 31));
}
__device__ __forceinline__ float decodeM(float raw) {
    unsigned u = __float_as_uint(raw);
    unsigned mask = 0x80000000u | (unsigned)((int)(~u) >> 31);
    return __uint_as_float(u ^ mask);
}
#define M_INIT_BITS 0x007FFFFFu   // encodeM(-inf)

// ---------------------------------------------------------------- kernels ---

// x[i,:] = emb[h[i],:] + pos_enc[i,:] @ Wp + bp
__global__ void k_embed(const int* __restrict__ hh, const float* __restrict__ pe,
                        const float* __restrict__ emb, const float* __restrict__ Wp,
                        const float* __restrict__ bp, float* __restrict__ x) {
    int i = blockIdx.x * blockDim.x + threadIdx.x;
    if (i >= NN) return;
    int t = hh[i];
    float p[PED];
#pragma unroll
    for (int j = 0; j < PED; ++j) p[j] = pe[i * PED + j];
#pragma unroll
    for (int c = 0; c < H; ++c) {
        float acc = emb[t * H + c] + bp[c];
#pragma unroll
        for (int j = 0; j < PED; ++j) acc += p[j] * Wp[j * H + c];
        x[i * H + c] = acc;
    }
}

// per-layer init (float4 granularity): agg=0, m=encode(-inf), den=0, stats=0
__global__ void k_init(float4* __restrict__ agg4, float4* __restrict__ m4,
                       float4* __restrict__ den4, float* __restrict__ stats) {
    int i = blockIdx.x * blockDim.x + threadIdx.x;
    float4 z = make_float4(0.f, 0.f, 0.f, 0.f);
    float mi = __uint_as_float(M_INIT_BITS);
    float4 mz = make_float4(mi, mi, mi, mi);
    if (i < NN * 4) agg4[i] = z;                 // NN*H floats
    if (i < NN * 2) { m4[i] = mz; den4[i] = z; } // NN*NHD floats each
    if (i < 64) stats[i] = 0.0f;
}

// Q/K/V projections via WMMA. One wave = one 16-row tile; 256 thr = 128 rows.
__global__ void k_qkv(const float* __restrict__ x,
                      const float* __restrict__ Wq, const float* __restrict__ bq,
                      const float* __restrict__ Wk, const float* __restrict__ bk,
                      const float* __restrict__ Wv, const float* __restrict__ bv,
                      float* __restrict__ q, float* __restrict__ k, float* __restrict__ v) {
    __shared__ __align__(32) _Float16 sBq[16 * 32], sBk[16 * 32], sBv[16 * 32];
    __shared__ float sbq[H], sbk[H], sbv[H];
    int tid = threadIdx.x;
    for (int idx = tid; idx < 512; idx += 256) {
        int col = idx >> 5, kk = idx & 31;
        _Float16 zq = (_Float16)0.f, zk = (_Float16)0.f, zv = (_Float16)0.f;
        if (kk < H) {
            zq = (_Float16)Wq[kk * H + col];
            zk = (_Float16)Wk[kk * H + col];
            zv = (_Float16)Wv[kk * H + col];
        }
        sBq[idx] = zq; sBk[idx] = zk; sBv[idx] = zv;
    }
    if (tid < H) { sbq[tid] = bq[tid]; sbk[tid] = bk[tid]; sbv[tid] = bv[tid]; }
    __syncthreads();

    int lane = tid & 31, wave = tid >> 5;
    int r = lane & 15, half = lane >> 4;
    int base = blockIdx.x * 128 + wave * 16;
    int node = base + r;
    const bool full = (int)(blockIdx.x + 1) * 128 <= NN;   // scalar (SGPR) guard

    v16h a = makeA_k16(x + (size_t)node * H + half * 8, node < NN);

    v8f dq = wmma16(a, loadB(sBq, r, half), bcastC(sbq[r]));
    v8f dk = wmma16(a, loadB(sBk, r, half), bcastC(sbk[r]));
    v8f dv = wmma16(a, loadB(sBv, r, half), bcastC(sbv[r]));

    const float qscale = 0.70710678f; // HD^-0.5
    size_t rowoff = (size_t)(base + half * 8) * H + r;
    float* qp = q + rowoff; float* kp = k + rowoff; float* vp = v + rowoff;
    if (full) {
#pragma unroll
        for (int i = 0; i < 8; ++i) {
            qp[i * H] = dq[i] * qscale;
            kp[i * H] = dk[i];
            vp[i * H] = dv[i];
        }
    } else {
#pragma unroll
        for (int i = 0; i < 8; ++i) {
            if (base + half * 8 + i < NN) {
                qp[i * H] = dq[i] * qscale;
                kp[i * H] = dk[i];
                vp[i * H] = dv[i];
            }
        }
    }
}

// per-edge per-head scores + segment max (by src) via branchless u32 atomic max
__global__ void k_sddmm(const int* __restrict__ ei, const float* __restrict__ q,
                        const float* __restrict__ k, float* __restrict__ logits,
                        float* __restrict__ m) {
    int e = blockIdx.x * blockDim.x + threadIdx.x;
    if (e >= EE) return;
    int s = ei[e], d = ei[EE + e];
    const float4* qp = (const float4*)(q + (size_t)s * H);
    const float4* kp = (const float4*)(k + (size_t)d * H);
    float4 q0 = qp[0], q1 = qp[1], q2 = qp[2], q3 = qp[3];
    float4 k0 = kp[0], k1 = kp[1], k2 = kp[2], k3 = kp[3];
    float l[8];
    l[0] = q0.x * k0.x + q0.y * k0.y;  l[1] = q0.z * k0.z + q0.w * k0.w;
    l[2] = q1.x * k1.x + q1.y * k1.y;  l[3] = q1.z * k1.z + q1.w * k1.w;
    l[4] = q2.x * k2.x + q2.y * k2.y;  l[5] = q2.z * k2.z + q2.w * k2.w;
    l[6] = q3.x * k3.x + q3.y * k3.y;  l[7] = q3.z * k3.z + q3.w * k3.w;
    float4* lp = (float4*)(logits + (size_t)e * NHD);
    lp[0] = make_float4(l[0], l[1], l[2], l[3]);
    lp[1] = make_float4(l[4], l[5], l[6], l[7]);
    unsigned* mr = (unsigned*)(m + (size_t)s * NHD);
#pragma unroll
    for (int hh = 0; hh < NHD; ++hh) atomicMax(mr + hh, encodeM(l[hh]));
}

// den[src,h] += exp(logit - m[src,h])
__global__ void k_den(const int* __restrict__ ei, const float* __restrict__ logits,
                      const float* __restrict__ m, float* __restrict__ den) {
    int e = blockIdx.x * blockDim.x + threadIdx.x;
    if (e >= EE) return;
    int s = ei[e];
    const float4* lp = (const float4*)(logits + (size_t)e * NHD);
    const float4* mp = (const float4*)(m + (size_t)s * NHD);
    float4 l0 = lp[0], l1 = lp[1], m0 = mp[0], m1 = mp[1];
    float ex[8];
    ex[0] = __expf(l0.x - decodeM(m0.x)); ex[1] = __expf(l0.y - decodeM(m0.y));
    ex[2] = __expf(l0.z - decodeM(m0.z)); ex[3] = __expf(l0.w - decodeM(m0.w));
    ex[4] = __expf(l1.x - decodeM(m1.x)); ex[5] = __expf(l1.y - decodeM(m1.y));
    ex[6] = __expf(l1.z - decodeM(m1.z)); ex[7] = __expf(l1.w - decodeM(m1.w));
    float* dr = den + (size_t)s * NHD;
#pragma unroll
    for (int hh = 0; hh < NHD; ++hh) atomicAdd(dr + hh, ex[hh]);
}

// agg[src,h,d] += softmax * v[dst,h,d]
__global__ void k_agg(const int* __restrict__ ei, const float* __restrict__ logits,
                      const float* __restrict__ m, const float* __restrict__ den,
                      const float* __restrict__ v, float* __restrict__ agg) {
    int e = blockIdx.x * blockDim.x + threadIdx.x;
    if (e >= EE) return;
    int s = ei[e], d = ei[EE + e];
    const float4* lp = (const float4*)(logits + (size_t)e * NHD);
    const float4* mp = (const float4*)(m + (size_t)s * NHD);
    const float4* dp = (const float4*)(den + (size_t)s * NHD);
    float4 l0 = lp[0], l1 = lp[1], m0 = mp[0], m1 = mp[1], d0 = dp[0], d1 = dp[1];
    float at[8];
    at[0] = __expf(l0.x - decodeM(m0.x)) / d0.x; at[1] = __expf(l0.y - decodeM(m0.y)) / d0.y;
    at[2] = __expf(l0.z - decodeM(m0.z)) / d0.z; at[3] = __expf(l0.w - decodeM(m0.w)) / d0.w;
    at[4] = __expf(l1.x - decodeM(m1.x)) / d1.x; at[5] = __expf(l1.y - decodeM(m1.y)) / d1.y;
    at[6] = __expf(l1.z - decodeM(m1.z)) / d1.z; at[7] = __expf(l1.w - decodeM(m1.w)) / d1.w;
    const float* vr = v + (size_t)d * H;
    float* ar = agg + (size_t)s * H;
#pragma unroll
    for (int hh = 0; hh < NHD; ++hh) {
        atomicAdd(ar + 2 * hh,     at[hh] * vr[2 * hh]);
        atomicAdd(ar + 2 * hh + 1, at[hh] * vr[2 * hh + 1]);
    }
}

// o = agg @ Wo + bo ; r = x + o ; accumulate BN1 column stats
__global__ void k_oproj(const float* __restrict__ agg, const float* __restrict__ x,
                        const float* __restrict__ Wo, const float* __restrict__ bo,
                        float* __restrict__ rbuf, float* __restrict__ stats) {
    __shared__ __align__(32) _Float16 sBo[16 * 32];
    __shared__ float sbo[H], sSum[H], sSsq[H];
    int tid = threadIdx.x;
    for (int idx = tid; idx < 512; idx += 256) {
        int col = idx >> 5, kk = idx & 31;
        sBo[idx] = (kk < H) ? (_Float16)Wo[kk * H + col] : (_Float16)0.f;
    }
    if (tid < H) { sbo[tid] = bo[tid]; sSum[tid] = 0.0f; sSsq[tid] = 0.0f; }
    __syncthreads();

    int lane = tid & 31, wave = tid >> 5;
    int r = lane & 15, half = lane >> 4;
    int base = blockIdx.x * 128 + wave * 16;
    int node = base + r;
    const bool full = (int)(blockIdx.x + 1) * 128 <= NN;

    v16h a = makeA_k16(agg + (size_t)node * H + half * 8, node < NN);
    v8f dD = wmma16(a, loadB(sBo, r, half), bcastC(sbo[r]));

    size_t rowoff = (size_t)(base + half * 8) * H + r;
    const float* xp = x + rowoff;
    float* rp = rbuf + rowoff;
    float ps = 0.0f, pq = 0.0f;
    if (full) {
#pragma unroll
        for (int i = 0; i < 8; ++i) {
            float rv = xp[i * H] + dD[i];
            rp[i * H] = rv;
            ps += rv; pq += rv * rv;
        }
    } else {
#pragma unroll
        for (int i = 0; i < 8; ++i) {
            if (base + half * 8 + i < NN) {
                float rv = xp[i * H] + dD[i];
                rp[i * H] = rv;
                ps += rv; pq += rv * rv;
            }
        }
    }
    atomicAdd(&sSum[r], ps);
    atomicAdd(&sSsq[r], pq);
    __syncthreads();
    if (tid < H) {
        atomicAdd(stats + tid, sSum[tid]);
        atomicAdd(stats + H + tid, sSsq[tid]);
    }
}

// xb = BN1(r) ; y = relu(xb@W1+b1)@W2+b2 ; r2 = xb + y ; BN2 stats
__global__ void k_ffn(const float* __restrict__ rbuf, const float* __restrict__ stats,
                      const float* __restrict__ g1, const float* __restrict__ be1,
                      const float* __restrict__ W1, const float* __restrict__ b1,
                      const float* __restrict__ W2, const float* __restrict__ b2,
                      float* __restrict__ r2buf, float* __restrict__ stats2) {
    __shared__ __align__(32) _Float16 sB1[32 * 32];   // 32 cols x 32 K (K<16 real)
    __shared__ __align__(32) _Float16 sB2[16 * 32];   // 16 cols x 32 K (K=32 real)
    __shared__ __align__(32) _Float16 tS[8 * 16 * 32]; // relu acts, per-wave 16x32
    __shared__ float sb1[2 * H], sb2[H];
    __shared__ float sScale[H], sShift[H], sSum[H], sSsq[H];
    int tid = threadIdx.x;
    for (int idx = tid; idx < 1024; idx += 256) {
        int col = idx >> 5, kk = idx & 31;
        sB1[idx] = (kk < H) ? (_Float16)W1[kk * 2 * H + col] : (_Float16)0.f;
    }
    for (int idx = tid; idx < 512; idx += 256) {
        int col = idx >> 5, kk = idx & 31;
        sB2[idx] = (_Float16)W2[kk * H + col];
    }
    if (tid < 2 * H) sb1[tid] = b1[tid];
    if (tid < H) {
        sb2[tid] = b2[tid];
        float mu  = stats[tid] * (1.0f / NN);
        float var = stats[H + tid] * (1.0f / NN) - mu * mu;
        float rs  = rsqrtf(var + EPSV) * g1[tid];
        sScale[tid] = rs;
        sShift[tid] = be1[tid] - mu * rs;
        sSum[tid] = 0.0f; sSsq[tid] = 0.0f;
    }
    __syncthreads();

    int lane = tid & 31, wave = tid >> 5;
    int r = lane & 15, half = lane >> 4;
    int base = blockIdx.x * 128 + wave * 16;
    int node = base + r;
    const bool full = (int)(blockIdx.x + 1) * 128 <= NN;

    // A = BN1-normalized x tile (K=16 padded to 32)
    v16h a;
#pragma unroll
    for (int j = 0; j < 16; ++j) a[j] = (_Float16)0.0f;
    if (node < NN) {
        const float4* p = (const float4*)(rbuf + (size_t)node * H + half * 8);
        float4 p0 = p[0], p1 = p[1];
        float rw[8] = {p0.x, p0.y, p0.z, p0.w, p1.x, p1.y, p1.z, p1.w};
#pragma unroll
        for (int j = 0; j < 8; ++j) {
            int c = half * 8 + j;
            a[j] = (_Float16)(rw[j] * sScale[c] + sShift[c]);
        }
    }

    // first FFN layer: N=32 -> two WMMAs over N halves
    v8f d1a = wmma16(a, loadB(sB1, r,      half), bcastC(sb1[r]));
    v8f d1b = wmma16(a, loadB(sB1, 16 + r, half), bcastC(sb1[16 + r]));

    _Float16* tw = tS + wave * 16 * 32;
#pragma unroll
    for (int i = 0; i < 8; ++i) {
        int row = half * 8 + i;
        tw[row * 32 + r]      = (_Float16)fmaxf(d1a[i], 0.0f);
        tw[row * 32 + 16 + r] = (_Float16)fmaxf(d1b[i], 0.0f);
    }
    __syncthreads();

    // A2 = relu activations, K=32 exact: two contiguous 16B LDS reads
    v8h lo = *(const v8h*)(tw + r * 32 + half * 8);
    v8h hi = *(const v8h*)(tw + r * 32 + 16 + half * 8);
    v16h a2 = __builtin_shufflevector(lo, hi, 0, 1, 2, 3, 4, 5, 6, 7,
                                      8, 9, 10, 11, 12, 13, 14, 15);
    v8f d2 = wmma16(a2, loadB(sB2, r, half), bcastC(sb2[r]));

    size_t rowoff = (size_t)(base + half * 8) * H + r;
    const float* rrow = rbuf + rowoff;
    float* r2p = r2buf + rowoff;
    float scl = sScale[r], shf = sShift[r];
    float ps = 0.0f, pq = 0.0f;
    if (full) {
#pragma unroll
        for (int i = 0; i < 8; ++i) {
            float xb = rrow[i * H] * scl + shf;
            float outv = xb + d2[i];
            r2p[i * H] = outv;
            ps += outv; pq += outv * outv;
        }
    } else {
#pragma unroll
        for (int i = 0; i < 8; ++i) {
            if (base + half * 8 + i < NN) {
                float xb = rrow[i * H] * scl + shf;
                float outv = xb + d2[i];
                r2p[i * H] = outv;
                ps += outv; pq += outv * outv;
            }
        }
    }
    atomicAdd(&sSum[r], ps);
    atomicAdd(&sSsq[r], pq);
    __syncthreads();
    if (tid < H) {
        atomicAdd(stats2 + tid, sSum[tid]);
        atomicAdd(stats2 + H + tid, sSsq[tid]);
    }
}

// x = BN2(r2), float4 granularity (4 consecutive columns per thread)
__global__ void k_bn2(const float4* __restrict__ r24, const float* __restrict__ stats2,
                      const float* __restrict__ g2, const float* __restrict__ be2,
                      float4* __restrict__ x4) {
    int i = blockIdx.x * blockDim.x + threadIdx.x;
    if (i >= NN * 4) return;                  // NN*H/4 float4s
    int c0 = (i & 3) * 4;                     // column base (0,4,8,12)
    float4 rv = r24[i];
    float o[4];
    float in[4] = {rv.x, rv.y, rv.z, rv.w};
#pragma unroll
    for (int j = 0; j < 4; ++j) {
        int c = c0 + j;
        float mu  = stats2[c] * (1.0f / NN);
        float var = stats2[H + c] * (1.0f / NN) - mu * mu;
        float rs  = rsqrtf(var + EPSV) * g2[c];
        o[j] = (in[j] - mu) * rs + be2[c];
    }
    x4[i] = make_float4(o[0], o[1], o[2], o[3]);
}

__global__ void k_pool_init(float* __restrict__ hg, float* __restrict__ cnt) {
    int i = blockIdx.x * blockDim.x + threadIdx.x;
    if (i < GG * H) hg[i] = 0.0f;
    if (i < GG) cnt[i] = 0.0f;
}

__global__ void k_pool(const float* __restrict__ x, const int* __restrict__ n2g,
                       float* __restrict__ hg, float* __restrict__ cnt) {
    int i = blockIdx.x * blockDim.x + threadIdx.x;
    if (i >= NN) return;
    int g = n2g[i];
    atomicAdd(cnt + g, 1.0f);
#pragma unroll
    for (int c = 0; c < H; ++c) atomicAdd(hg + g * H + c, x[(size_t)i * H + c]);
}

__global__ void k_head(const float* __restrict__ hg, const float* __restrict__ cnt,
                       const float* __restrict__ Wc1, const float* __restrict__ bc1,
                       const float* __restrict__ Wc2, const float* __restrict__ bc2,
                       const float* __restrict__ Wc3, const float* __restrict__ bc3,
                       float* __restrict__ out) {
    int g = threadIdx.x;
    if (g >= GG) return;
    float inv = 1.0f / cnt[g];
    float a[H];
#pragma unroll
    for (int c = 0; c < H; ++c) a[c] = hg[g * H + c] * inv;
    float h1[8];
#pragma unroll
    for (int j = 0; j < 8; ++j) {
        float s = bc1[j];
#pragma unroll
        for (int c = 0; c < H; ++c) s += a[c] * Wc1[c * 8 + j];
        h1[j] = fmaxf(s, 0.0f);
    }
    float h2[4];
#pragma unroll
    for (int j = 0; j < 4; ++j) {
        float s = bc2[j];
#pragma unroll
        for (int c = 0; c < 8; ++c) s += h1[c] * Wc2[c * 4 + j];
        h2[j] = fmaxf(s, 0.0f);
    }
    float s = bc3[0];
#pragma unroll
    for (int c = 0; c < 4; ++c) s += h2[c] * Wc3[c];
    out[g] = s;
}

// ----------------------------------------------------------------- launch ---

extern "C" void kernel_launch(void* const* d_in, const int* in_sizes, int n_in,
                              void* d_out, int out_size, void* d_ws, size_t ws_size,
                              hipStream_t stream) {
    const int*   hIdx = (const int*)  d_in[0];
    const float* pe   = (const float*)d_in[1];
    const int*   ei   = (const int*)  d_in[2];
    const int*   n2g  = (const int*)  d_in[3];
    const float* emb  = (const float*)d_in[4];
    const float* Wp = (const float*)d_in[5];  const float* bp  = (const float*)d_in[6];
    const float* Wq = (const float*)d_in[7];  const float* bq  = (const float*)d_in[8];
    const float* Wk = (const float*)d_in[9];  const float* bk  = (const float*)d_in[10];
    const float* Wv = (const float*)d_in[11]; const float* bv  = (const float*)d_in[12];
    const float* Wo = (const float*)d_in[13]; const float* bo  = (const float*)d_in[14];
    const float* g1 = (const float*)d_in[15]; const float* be1 = (const float*)d_in[16];
    const float* W1 = (const float*)d_in[17]; const float* bf1 = (const float*)d_in[18];
    const float* W2 = (const float*)d_in[19]; const float* bf2 = (const float*)d_in[20];
    const float* g2 = (const float*)d_in[21]; const float* be2 = (const float*)d_in[22];
    const float* Wc1 = (const float*)d_in[23]; const float* bc1 = (const float*)d_in[24];
    const float* Wc2 = (const float*)d_in[25]; const float* bc2 = (const float*)d_in[26];
    const float* Wc3 = (const float*)d_in[27]; const float* bc3 = (const float*)d_in[28];
    float* out = (float*)d_out;

    float* ws = (float*)d_ws;
    size_t o = 0;
    float* x    = ws + o; o += (size_t)NN * H;
    float* qb   = ws + o; o += (size_t)NN * H;   // reused as BN1 residual buffer
    float* kb   = ws + o; o += (size_t)NN * H;   // reused as r2 buffer
    float* vb   = ws + o; o += (size_t)NN * H;
    float* agg  = ws + o; o += (size_t)NN * H;
    float* mb   = ws + o; o += (size_t)NN * NHD; // encoded-u32 segment max
    float* den  = ws + o; o += (size_t)NN * NHD;
    float* lg   = ws + o; o += (size_t)EE * NHD;
    float* st   = ws + o; o += 64;               // [0:16)sum1 [16:32)ssq1 [32:48)sum2 [48:64)ssq2
    float* hg   = ws + o; o += (size_t)GG * H;
    float* cnt  = ws + o; o += (size_t)GG;

    const int NB_W   = (NN + 127) / 128;          // wmma tile blocks
    const int NB_E   = (EE + 255) / 256;          // edge blocks
    const int NB_NH4 = (NN * 4 + 255) / 256;      // float4 elementwise blocks
    const int NB_N   = (NN + 255) / 256;

    k_embed<<<NB_N, 256, 0, stream>>>(hIdx, pe, emb, Wp, bp, x);

    for (int l = 0; l < LL; ++l) {
        const float* wq = Wq + (size_t)l * H * H;   const float* bql = bq + (size_t)l * H;
        const float* wk = Wk + (size_t)l * H * H;   const float* bkl = bk + (size_t)l * H;
        const float* wv = Wv + (size_t)l * H * H;   const float* bvl = bv + (size_t)l * H;
        const float* wo = Wo + (size_t)l * H * H;   const float* bol = bo + (size_t)l * H;
        const float* w1 = W1 + (size_t)l * H * 2 * H; const float* b1l = bf1 + (size_t)l * 2 * H;
        const float* w2 = W2 + (size_t)l * 2 * H * H; const float* b2l = bf2 + (size_t)l * H;
        const float* g1l = g1 + (size_t)l * H;  const float* be1l = be1 + (size_t)l * H;
        const float* g2l = g2 + (size_t)l * H;  const float* be2l = be2 + (size_t)l * H;

        k_init <<<NB_NH4, 256, 0, stream>>>((float4*)agg, (float4*)mb, (float4*)den, st);
        k_qkv  <<<NB_W, 256, 0, stream>>>(x, wq, bql, wk, bkl, wv, bvl, qb, kb, vb);
        k_sddmm<<<NB_E, 256, 0, stream>>>(ei, qb, kb, lg, mb);
        k_den  <<<NB_E, 256, 0, stream>>>(ei, lg, mb, den);
        k_agg  <<<NB_E, 256, 0, stream>>>(ei, lg, mb, den, vb, agg);
        k_oproj<<<NB_W, 256, 0, stream>>>(agg, x, wo, bol, qb, st);           // r -> qb
        k_ffn  <<<NB_W, 256, 0, stream>>>(qb, st, g1l, be1l, w1, b1l, w2, b2l,
                                          kb, st + 32);                        // r2 -> kb
        k_bn2  <<<NB_NH4, 256, 0, stream>>>((const float4*)kb, st + 32, g2l, be2l,
                                            (float4*)x);
    }

    k_pool_init<<<(GG * H + 255) / 256, 256, 0, stream>>>(hg, cnt);
    k_pool     <<<NB_N, 256, 0, stream>>>(x, n2g, hg, cnt);
    k_head     <<<1, 128, 0, stream>>>(hg, cnt, Wc1, bc1, Wc2, bc2, Wc3, bc3, out);
}
